// DiffRenderer_49134425866425
// MI455X (gfx1250) — matrix operand order repaired and checked by reading, not compiled, not run on hardware
//
#include <hip/hip_runtime.h>
#include <hip/hip_bf16.h>

typedef __attribute__((ext_vector_type(16))) _Float16 v16h;
typedef __attribute__((ext_vector_type(8)))  _Float16 v8h;
typedef __attribute__((ext_vector_type(8)))  float    v8f;
typedef __attribute__((ext_vector_type(4)))  int      v4i;

#define ROWS   384
#define COLS   768
#define NCH    69
#define KPAD   96          // 3 x K32 WMMA steps
#define CH     28
#define CW     14
#define NPIX   392         // CH*CW
#define NPAD   400         // 25 x N16 tiles
#define IMGW   (COLS * CW) // 10752
#define CELLS_PER_BLOCK 128
#define NTILES 25
#define FONT_BYTES (NPAD * KPAD * 2)   // 76800
#define SHMEM_BYTES ((NPAD * KPAD + CELLS_PER_BLOCK * KPAD) * 2)  // 101376

// ---------------------------------------------------------------------------
// Kernel 1: pack font (69,28,14) f32 -> fontB[n=0..399][k=0..95] f16 in d_ws.
// ---------------------------------------------------------------------------
__global__ void prep_font_kernel(const float* __restrict__ font,
                                 _Float16* __restrict__ fontB) {
    int i = blockIdx.x * 256 + threadIdx.x;   // over NPAD*KPAD
    if (i >= NPAD * KPAD) return;
    int n = i / KPAD;
    int k = i - n * KPAD;
    float v = 0.0f;
    if (k < NCH && n < NPIX) v = font[k * NPIX + n];
    fontB[i] = (_Float16)v;
}

// B fragment (32x16 f16 layout): lane = n, halfs 0..7 = K(koff..koff+7),
// halfs 8..15 = K(koff+16..+23), koff = kt*32 + half*8.
__device__ __forceinline__ v16h load_bfrag(const _Float16* sFont, int p,
                                           int kt, int half) {
    const _Float16* bp = sFont + p * KPAD + kt * 32 + half * 8;
    v8h lo = *(const v8h*)(bp + 0);
    v8h hi = *(const v8h*)(bp + 16);
    return __builtin_shufflevector(lo, hi, 0,1,2,3,4,5,6,7,8,9,10,11,12,13,14,15);
}

// ---------------------------------------------------------------------------
// Kernel 2: fused softmax + WMMA blend + image assembly.
// Block: 256 threads (8 wave32), 128 cells. Each wave owns one 16-cell M-tile.
// ---------------------------------------------------------------------------
__global__ void __launch_bounds__(256)
diff_render_kernel(const float* __restrict__ logits,
                   const _Float16* __restrict__ fontB,
                   float* __restrict__ out) {
    extern __shared__ __align__(32) char smem[];
    _Float16* sFont = (_Float16*)smem;                  // 76800 B
    _Float16* sProb = (_Float16*)(smem + FONT_BYTES);   // 24576 B

    const int tid = threadIdx.x;
    const int cellBlockBase = blockIdx.x * CELLS_PER_BLOCK;

    if (tid < CELLS_PER_BLOCK) {
        // ---- softmax for cell (cellBlockBase + tid), staged in LDS ----
        const float* lp = logits + (size_t)(cellBlockBase + tid) * NCH;
        _Float16* pr = sProb + tid * KPAD;
        float m = -3.0e38f;
        for (int k = 0; k < NCH; ++k) {                 // single global pass (NT)
            float x = __builtin_nontemporal_load(lp + k);
            pr[k] = (_Float16)x;                        // stash logit as f16
            m = fmaxf(m, x);
        }
        float s = 0.0f;
        for (int k = 0; k < NCH; ++k) {
            float e = __expf((float)pr[k] - m);
            s += e;
            pr[k] = (_Float16)e;
        }
        float inv = __frcp_rn(s);
        for (int k = 0; k < NCH; ++k)
            pr[k] = (_Float16)((float)pr[k] * inv);
        for (int k = NCH; k < KPAD; ++k)
            pr[k] = (_Float16)0.0f;
    } else {
        // ---- other half of the block brings packed font into LDS ----
        const int nChunks = FONT_BYTES / 16;            // 4800 x 16B
#if __has_builtin(__builtin_amdgcn_global_load_async_to_lds_b128)
        typedef __attribute__((address_space(1))) v4i g_v4i;
        typedef __attribute__((address_space(3))) v4i lds_v4i;
        g_v4i*   gsrc = (g_v4i*)fontB;
        lds_v4i* ldst = (lds_v4i*)sFont;
        for (int i = tid - 128; i < nChunks; i += 128) {
            __builtin_amdgcn_global_load_async_to_lds_b128(gsrc + i, ldst + i, 0, 0);
        }
#if __has_builtin(__builtin_amdgcn_s_wait_asynccnt)
        __builtin_amdgcn_s_wait_asynccnt(0);
#else
        asm volatile("s_wait_asynccnt 0x0" ::: "memory");
#endif
#else
        const uint4* src = (const uint4*)fontB;
        uint4* dst = (uint4*)sFont;
        for (int i = tid - 128; i < nChunks; i += 128)
            dst[i] = src[i];
#endif
    }
    __syncthreads();

    const int wave = tid >> 5;
    const int lane = tid & 31;
    const int half = lane >> 4;   // 0 or 1
    const int l16  = lane & 15;
    const int mbase = wave * 16;

    // ---- A fragments (16x32 f16 layout): lane holds row M=l16,
    //      halfs 0..7 = K(koff..koff+7), halfs 8..15 = K(koff+16..+23),
    //      koff = ktile*32 + half*8 ----
    v16h a0, a1, a2;
    {
        const _Float16* base = sProb + (mbase + l16) * KPAD + half * 8;
        v8h lo, hi;
        lo = *(const v8h*)(base + 0);   hi = *(const v8h*)(base + 16);
        a0 = __builtin_shufflevector(lo, hi, 0,1,2,3,4,5,6,7,8,9,10,11,12,13,14,15);
        lo = *(const v8h*)(base + 32);  hi = *(const v8h*)(base + 48);
        a1 = __builtin_shufflevector(lo, hi, 0,1,2,3,4,5,6,7,8,9,10,11,12,13,14,15);
        lo = *(const v8h*)(base + 64);  hi = *(const v8h*)(base + 80);
        a2 = __builtin_shufflevector(lo, hi, 0,1,2,3,4,5,6,7,8,9,10,11,12,13,14,15);
    }

    // ---- per-lane output row bases for the 8 D rows (M = j + 8*half) ----
    int outBase[8];
#pragma unroll
    for (int j = 0; j < 8; ++j) {
        int cell = cellBlockBase + mbase + j + 8 * half;
        int r = cell / COLS;
        int c = cell - r * COLS;
        outBase[j] = r * (CH * IMGW) + c * CW;
    }

    // ---- software-pipelined: 25 N-tiles x 3 K-steps of wmma_f32_16x16x32_f16,
    //      tile 24 peeled (only tile with partially valid pixels) ----
    v16h b0 = load_bfrag(sFont, l16, 0, half);
    v16h b1 = load_bfrag(sFont, l16, 1, half);
    v16h b2 = load_bfrag(sFont, l16, 2, half);

    for (int t = 0; t < NTILES - 1; ++t) {
        int pn = (t + 1) * 16 + l16;                    // prefetch next tile
        v16h n0 = load_bfrag(sFont, pn, 0, half);
        v16h n1 = load_bfrag(sFont, pn, 1, half);
        v16h n2 = load_bfrag(sFont, pn, 2, half);

        v8f acc = {};
        acc = __builtin_amdgcn_wmma_f32_16x16x32_f16(false, a0, false, b0,
                                                     (short)0, acc, false, false);
        acc = __builtin_amdgcn_wmma_f32_16x16x32_f16(false, a1, false, b1,
                                                     (short)0, acc, false, false);
        acc = __builtin_amdgcn_wmma_f32_16x16x32_f16(false, a2, false, b2,
                                                     (short)0, acc, false, false);

        int p = t * 16 + l16;
        int h = p / CW;
        int w = p - h * CW;
        int off = h * IMGW + w;
#pragma unroll
        for (int j = 0; j < 8; ++j)
            __builtin_nontemporal_store(acc[j], &out[outBase[j] + off]);

        b0 = n0; b1 = n1; b2 = n2;
    }

    // tail tile (t = 24): pixels 384+l16, lanes with p >= 392 are padding
    {
        v8f acc = {};
        acc = __builtin_amdgcn_wmma_f32_16x16x32_f16(false, a0, false, b0,
                                                     (short)0, acc, false, false);
        acc = __builtin_amdgcn_wmma_f32_16x16x32_f16(false, a1, false, b1,
                                                     (short)0, acc, false, false);
        acc = __builtin_amdgcn_wmma_f32_16x16x32_f16(false, a2, false, b2,
                                                     (short)0, acc, false, false);
        int p = (NTILES - 1) * 16 + l16;
        if (p < NPIX) {
            int h = p / CW;
            int w = p - h * CW;
            int off = h * IMGW + w;
#pragma unroll
            for (int j = 0; j < 8; ++j)
                __builtin_nontemporal_store(acc[j], &out[outBase[j] + off]);
        }
    }
}

extern "C" void kernel_launch(void* const* d_in, const int* in_sizes, int n_in,
                              void* d_out, int out_size, void* d_ws, size_t ws_size,
                              hipStream_t stream) {
    const float* logits = (const float*)d_in[0];   // (384,768,69) f32
    const float* font   = (const float*)d_in[1];   // (69,28,14)  f32
    float* out          = (float*)d_out;           // (10752,10752) f32
    _Float16* fontB     = (_Float16*)d_ws;         // 400*96 f16 = 76800 B

    // allow >64KB dynamic LDS on gfx1250 (320KB WGP LDS)
    (void)hipFuncSetAttribute((const void*)diff_render_kernel,
                              hipFuncAttributeMaxDynamicSharedMemorySize,
                              SHMEM_BYTES);

    prep_font_kernel<<<(NPAD * KPAD + 255) / 256, 256, 0, stream>>>(font, fontB);

    const int nCells = ROWS * COLS;                       // 294912
    const int nBlocks = nCells / CELLS_PER_BLOCK;         // 2304
    diff_render_kernel<<<nBlocks, 256, (size_t)SHMEM_BYTES, stream>>>(logits, fontB, out);
}
